// DynamicsBackbone_71640054497688
// MI455X (gfx1250) — compile-verified, matrix operand order checked by reading, not run on hardware
//
#include <hip/hip_runtime.h>
#include <math.h>

typedef __attribute__((ext_vector_type(16))) _Float16 v16h;
typedef __attribute__((ext_vector_type(8)))  _Float16 v8h;
typedef __attribute__((ext_vector_type(8)))  float    v8f;
typedef __attribute__((ext_vector_type(4)))  unsigned int v4u;
typedef __attribute__((ext_vector_type(8)))  int      v8i;
typedef __attribute__((ext_vector_type(4)))  int      v4i;
typedef __attribute__((ext_vector_type(4)))  unsigned int uint4v;

#define BB   2
#define SS   8
#define NN   150
#define HH   128
#define FF   40
#define EFP  96     // EF=82 padded to 3*32
#define TM   32     // edge tile (M)
#define NTIL 5      // 160 padded edges / 32
#define NEDGE 149

// ---------------- WMMA LDS operand loaders (wave32, 16x16x32 f16) ----------
// A 16x32 layout: lane m=lane&15; halves 0..7 -> K = (lane<16?0:8)+0..7,
//                 halves 8..15 -> K = (lane<16?16:24)+0..7
__device__ __forceinline__ v16h lds_load_A(const _Float16* p0, int row0, int k0,
                                           int ld, int lane) {
  const int m  = lane & 15;
  const int kb = (lane < 16) ? 0 : 8;
  const _Float16* p = p0 + (row0 + m) * ld + k0 + kb;
  v8h lo = *(const v8h*)(p);        // 16B aligned -> ds_load_b128
  v8h hi = *(const v8h*)(p + 16);   // 16B aligned -> ds_load_b128
  v16h a;
#pragma unroll
  for (int i = 0; i < 8; ++i) { a[i] = lo[i]; a[i + 8] = hi[i]; }
  return a;
}
// B fragments pre-swizzled in LDS: [kt][nt][lane][16 halves] contiguous.
__device__ __forceinline__ v16h lds_load_Bswz(const _Float16* sWs, int kt, int nt,
                                              int lane) {
  return *(const v16h*)(sWs + ((((kt << 3) + nt) << 5) + lane) * 16);
}

// silu via fast reciprocal: v * rcp(1+exp(-v))  (avoids IEEE div expansion)
__device__ __forceinline__ float silu(float v) {
  return v * __builtin_amdgcn_rcpf(1.f + __expf(-v));
}

// ---------------- TDM 1-D contiguous copy global -> LDS --------------------
__device__ __forceinline__ void tdm_copy_to_lds(const _Float16* gsrc,
                                                _Float16* ldst, int bytes) {
#if __has_builtin(__builtin_amdgcn_tensor_load_to_lds)
  const unsigned units = (unsigned)bytes >> 3;        // 8-byte elements
  const unsigned lds_addr = (unsigned)(size_t)ldst;   // LDS aperture low bits
  const unsigned long long ga = (unsigned long long)(size_t)gsrc;
  v4u g0;
  g0[0] = 1u;                                   // count=1 (valid), user mode
  g0[1] = lds_addr;                             // lds_addr
  g0[2] = (unsigned)ga;                         // global_addr[31:0]
  g0[3] = (unsigned)((ga >> 32) & 0x01FFFFFFu)  // global_addr[56:32]
          | (2u << 30);                         // type=2 (image)
  v8i g1;
  g1[0] = (int)(3u << 16);                      // wg_mask=0, data_size=3 (8B)
  g1[1] = (int)((units & 0xFFFFu) << 16);       // tensor_dim0[15:0]
  g1[2] = (int)(((units >> 16) & 0xFFFFu) | (1u << 16)); // dim0 hi | dim1=1
  g1[3] = (int)((units & 0xFFFFu) << 16);       // tile_dim0 = units
  g1[4] = 0;                                    // tile_dim1=0, tile_dim2=0
  g1[5] = (int)units;                           // tensor_dim0_stride
  g1[6] = 0;
  g1[7] = 0;
  v4i z4 = {0, 0, 0, 0};
#if defined(__clang_major__) && (__clang_major__ >= 23)
  v8i z8 = {0, 0, 0, 0, 0, 0, 0, 0};
  __builtin_amdgcn_tensor_load_to_lds(g0, g1, z4, z4, z8, 0);
#else
  __builtin_amdgcn_tensor_load_to_lds(g0, g1, z4, z4, 0);
#endif
#else
  // Fallback: vector copy by wave 0 (32 lanes, 16B each)
  const uint4v* s = (const uint4v*)gsrc;
  uint4v* d = (uint4v*)ldst;
  const int n16 = bytes >> 4;
  for (int k = (int)(threadIdx.x & 31); k < n16; k += 32) d[k] = s[k];
#endif
}

__device__ __forceinline__ void tdm_wait() {
#if __has_builtin(__builtin_amdgcn_s_wait_tensorcnt)
  __builtin_amdgcn_s_wait_tensorcnt(0);
#endif
}

// ---------------- prep kernels ---------------------------------------------
__global__ void h0_kernel(const int* __restrict__ pep, const int* __restrict__ lab,
                          const int* __restrict__ aapos, float* __restrict__ hbuf) {
  int idx = blockIdx.x * blockDim.x + threadIdx.x;
  if (idx >= BB * NN) return;
  int b = idx / NN, n = idx % NN;
  int l   = lab[b * NN + n];
  int pos = aapos[b * NN + n] - 1;
  int aa  = pep[b * 15 + pos];
  float emb[FF];
#pragma unroll
  for (int c = 0; c < FF; ++c) emb[c] = 0.f;
  if (l  >= 0 && l  < 5)  emb[l]        = 1.f;
  if (aa >= 0 && aa < 20) emb[5 + aa]   = 1.f;
  if (pos>= 0 && pos< 15) emb[25 + pos] = 1.f;
  for (int s = 0; s < SS; ++s)
    for (int c = 0; c < FF; ++c)
      hbuf[(((b * SS + s) * NN) + n) * FF + c] = emb[c];
}

// x_r[b,s,i,d] = sum_j x[b,s,j,d] * R[b,i,j];  also writes x0 copy
__global__ void restore_kernel(const float* __restrict__ x, const float* __restrict__ R,
                               float* __restrict__ xr, float* __restrict__ x0) {
  int idx = blockIdx.x * blockDim.x + threadIdx.x;
  if (idx >= BB * SS * NN * 3) return;
  int d = idx % 3;
  int i = (idx / 3) % NN;
  int s = (idx / (3 * NN)) % SS;
  int b = idx / (3 * NN * SS);
  const float* xs = x + ((size_t)(b * SS + s) * NN) * 3 + d;
  const float* Ri = R + ((size_t)b * NN + i) * NN;
  float acc = 0.f;
  for (int j = 0; j < NN; ++j) acc += xs[j * 3] * Ri[j];
  xr[idx] = acc;
  x0[idx] = acc;
}

// Pack fp32 [src_rows,128] weight into WMMA B-fragment-swizzled f16:
// layout [kt][nt][lane][h]; lane holds col n = nt*16+(lane&15),
// halves 0..7 -> K=(lane<16?0:8)+0..7, halves 8..15 -> +16. Zero-pad K.
__global__ void pack_swz_f16(const float* __restrict__ src, _Float16* __restrict__ dst,
                             int src_rows, int dst_rows) {
  int idx = blockIdx.x * blockDim.x + threadIdx.x;
  int total = dst_rows * HH;
  if (idx >= total) return;
  int h    = idx & 15;
  int lane = (idx >> 4) & 31;
  int nt   = (idx >> 9) & 7;
  int kt   = idx >> 12;
  int kb = (lane < 16) ? 0 : 8;
  int klocal = (h < 8) ? (kb + h) : (kb + 16 + (h - 8));
  int k = kt * 32 + klocal;
  int n = nt * 16 + (lane & 15);
  dst[idx] = (_Float16)((k < src_rows) ? src[k * HH + n] : 0.f);
}

// temb[b,h] = t[b]*w_t[h] + w_b_t[h]  (or 0 for tail layers)
__global__ void temb_kernel(const float* __restrict__ t, const float* __restrict__ wt,
                            const float* __restrict__ wbt, float* __restrict__ temb,
                            int has) {
  int idx = blockIdx.x * blockDim.x + threadIdx.x;
  if (idx >= BB * HH) return;
  int b = idx / HH, h = idx % HH;
  temb[idx] = has ? (t[b] * wt[h] + wbt[h]) : 0.f;
}

__global__ void final_kernel(const float* __restrict__ xf, const float* __restrict__ x0,
                             const float* __restrict__ am, float* __restrict__ out) {
  int idx = blockIdx.x * blockDim.x + threadIdx.x;
  if (idx >= BB * SS * NN * 3) return;
  int n = (idx / 3) % NN;
  int b = idx / (3 * NN * SS);
  out[idx] = (xf[idx] - x0[idx]) * am[b * NN + n];
}

// ---------------- fused EGNN layer: one workgroup per (b,s,i) --------------
__global__ __launch_bounds__(256)
void egnn_layer(const float* __restrict__ x_in, const float* __restrict__ h_in,
                float* __restrict__ x_out, float* __restrict__ h_out,
                const float* __restrict__ bond, const float* __restrict__ emask,
                const _Float16* __restrict__ We1p,  // swizzled [3kt][8nt][32][16]
                const _Float16* __restrict__ We2p,  // swizzled [4kt][8nt][32][16]
                const float* __restrict__ be1, const float* __restrict__ be2,
                const float* __restrict__ temb,     // [B][H]
                const float* __restrict__ Wc, const float* __restrict__ bc,
                const float* __restrict__ Wh, const float* __restrict__ bh) {
  __shared__ __align__(32) _Float16 sW[HH * HH];  // 32 KB: We1 then We2 (swz)
  __shared__ __align__(32) _Float16 sA[TM * HH];  // 8 KB : ef tile then m2
  __shared__ __align__(32) _Float16 sM[TM * HH];  // 8 KB : m1
  __shared__ float sDiff[TM][3];
  __shared__ float sEm[TM];
  __shared__ int   sJ[TM];
  __shared__ float sHi[FF];
  __shared__ float sAgg[HH];
  __shared__ float sXout[3];

  const int i = blockIdx.x % NN;
  const int s = (blockIdx.x / NN) % SS;
  const int b = blockIdx.x / (NN * SS);
  const int tid  = threadIdx.x;
  const int wave = tid >> 5;
  const int lane = tid & 31;

  const size_t node = (size_t)(b * SS + s) * NN + i;
  const float* xi = x_in + node * 3;
  const float* hi = h_in + node * FF;
  const float xi0 = xi[0], xi1 = xi[1], xi2 = xi[2];

  for (int c = tid; c < HH; c += 256) sAgg[c] = 0.f;
  if (tid < 3)  sXout[tid] = 0.f;
  if (tid < FF) sHi[tid] = hi[tid];
  __syncthreads();

  for (int tt = 0; tt < NTIL; ++tt) {
    // --- kick off async We1 DMA into sW (overlaps with ef build)
    if (wave == 0) tdm_copy_to_lds(We1p, sW, EFP * HH * 2);

    // --- per-edge geometry + ef tail columns
    if (tid < TM) {
      int t = tt * TM + tid;
      float d0 = 0.f, d1 = 0.f, d2 = 0.f, dist = 0.f, em = 0.f, bd = 0.f;
      int j = 0;
      if (t < NEDGE) {
        j = (t < i) ? t : t + 1;
        const float* xj = x_in + ((size_t)(b * SS + s) * NN + j) * 3;
        d0 = xi0 - xj[0]; d1 = xi1 - xj[1]; d2 = xi2 - xj[2];
        dist = sqrtf(d0 * d0 + d1 * d1 + d2 * d2 + 1e-12f);
        em = emask[((size_t)b * NN + i) * NN + j];
        bd = bond [((size_t)b * NN + i) * NN + j];
      }
      sDiff[tid][0] = d0; sDiff[tid][1] = d1; sDiff[tid][2] = d2;
      sEm[tid] = em; sJ[tid] = j;
      sA[tid * HH + 80] = (_Float16)dist;
      sA[tid * HH + 81] = (_Float16)bd;
#pragma unroll
      for (int c = 82; c < EFP; ++c) sA[tid * HH + c] = (_Float16)0.f;
    }
    __syncthreads();
    // --- ef tile h-part: cols 0..39 = h[i], 40..79 = h[j]
    for (int idx = tid; idx < TM * 80; idx += 256) {
      int r = idx / 80, c = idx % 80;
      int t = tt * TM + r;
      float v = 0.f;
      if (t < NEDGE) {
        if (c < FF) v = sHi[c];
        else        v = h_in[((size_t)(b * SS + s) * NN + sJ[r]) * FF + (c - FF)];
      }
      sA[r * HH + c] = (_Float16)v;
    }
    if (wave == 0) tdm_wait();
    __syncthreads();

    // --- GEMM1: m1 = silu(ef @ We1 + be1 + temb); 16 tiles / 8 waves
#pragma unroll
    for (int q = 0; q < 2; ++q) {
      const int mi = q, ni = wave;
      v8f acc = {};
#pragma unroll
      for (int kk = 0; kk < 3; ++kk) {
        v16h a = lds_load_A(sA, mi * 16, kk * 32, HH, lane);
        v16h w = lds_load_Bswz(sW, kk, ni, lane);
        acc = __builtin_amdgcn_wmma_f32_16x16x32_f16(false, a, false, w,
                                                     (short)0, acc, false, false);
      }
      const int n = lane & 15, mb = (lane < 16) ? 0 : 8;
      const int col = ni * 16 + n;
      const float bias = be1[col] + temb[b * HH + col];
#pragma unroll
      for (int v = 0; v < 8; ++v)
        sM[(mi * 16 + mb + v) * HH + col] = (_Float16)silu(acc[v] + bias);
    }
    __syncthreads();
    // --- async We2 DMA into sW
    if (wave == 0) { tdm_copy_to_lds(We2p, sW, HH * HH * 2); tdm_wait(); }
    __syncthreads();

    // --- GEMM2: m2 = silu(m1 @ We2 + be2) * em
#pragma unroll
    for (int q = 0; q < 2; ++q) {
      const int mi = q, ni = wave;
      v8f acc = {};
#pragma unroll
      for (int kk = 0; kk < 4; ++kk) {
        v16h a = lds_load_A(sM, mi * 16, kk * 32, HH, lane);
        v16h w = lds_load_Bswz(sW, kk, ni, lane);
        acc = __builtin_amdgcn_wmma_f32_16x16x32_f16(false, a, false, w,
                                                     (short)0, acc, false, false);
      }
      const int n = lane & 15, mb = (lane < 16) ? 0 : 8;
      const int col = ni * 16 + n;
      const float bias = be2[col];
#pragma unroll
      for (int v = 0; v < 8; ++v) {
        const int row = mi * 16 + mb + v;
        sA[row * HH + col] = (_Float16)(silu(acc[v] + bias) * sEm[row]);
      }
    }
    __syncthreads();

    // --- cw = m2 @ Wc + bc; accumulate x displacement (all edges -> node i)
    if (tid < TM) {
      float acc = 0.f;
      for (int c = 0; c < HH; ++c) acc += (float)sA[tid * HH + c] * Wc[c];
      const float cw = acc + bc[0];
      atomicAdd(&sXout[0], sDiff[tid][0] * cw);
      atomicAdd(&sXout[1], sDiff[tid][1] * cw);
      atomicAdd(&sXout[2], sDiff[tid][2] * cw);
    }
    // --- agg[i] += column sums of m2
    if (tid < HH) {
      float a = 0.f;
      for (int r = 0; r < TM; ++r) a += (float)sA[r * HH + tid];
      sAgg[tid] += a;
    }
    __syncthreads();
  }

  // --- node update: h' = silu(cat(h, agg) @ Wh + bh), x' = x + xout/(N-1)
  if (tid < FF) {
    float acc = bh[tid];
    for (int k = 0; k < FF; ++k) acc += sHi[k] * Wh[k * FF + tid];
    for (int k = 0; k < HH; ++k) acc += sAgg[k] * Wh[(FF + k) * FF + tid];
    h_out[node * FF + tid] = silu(acc);
  }
  if (tid < 3) {
    const float xv = (tid == 0) ? xi0 : (tid == 1) ? xi1 : xi2;
    x_out[node * 3 + tid] = xv + sXout[tid] * (1.f / (float)(NN - 1));
  }
}

// ---------------- host-side launch -----------------------------------------
extern "C" void kernel_launch(void* const* d_in, const int* in_sizes, int n_in,
                              void* d_out, int out_size, void* d_ws, size_t ws_size,
                              hipStream_t stream) {
  (void)in_sizes; (void)n_in; (void)out_size; (void)ws_size;
  const float* t_in   = (const float*)d_in[0];
  const float* x_in   = (const float*)d_in[1];
  const int*   pep    = (const int*)d_in[2];
  const int*   labels = (const int*)d_in[3];
  const int*   aapos  = (const int*)d_in[4];
  const float* bond   = (const float*)d_in[5];
  const float* emask  = (const float*)d_in[6];
  const float* amask  = (const float*)d_in[7];
  const float* Rmat   = (const float*)d_in[8];

  // workspace carve
  size_t off = 0;
  auto alloc = [&](size_t bytes) -> char* {
    char* p = (char*)d_ws + off;
    off += (bytes + 255) & ~(size_t)255;
    return p;
  };
  const size_t nx = (size_t)BB * SS * NN * 3;
  const size_t nh = (size_t)BB * SS * NN * FF;
  float* xb0 = (float*)alloc(nx * 4);
  float* xb1 = (float*)alloc(nx * 4);
  float* x0  = (float*)alloc(nx * 4);
  float* hb0 = (float*)alloc(nh * 4);
  float* hb1 = (float*)alloc(nh * 4);
  _Float16* We1p = (_Float16*)alloc((size_t)3 * EFP * HH * 2);
  _Float16* We2p = (_Float16*)alloc((size_t)3 * HH * HH * 2);
  float* temb = (float*)alloc((size_t)3 * BB * HH * 4);

  // prep: embedding + coordinate restore
  h0_kernel<<<(BB * NN + 255) / 256, 256, 0, stream>>>(pep, labels, aapos, hb0);
  restore_kernel<<<((int)nx + 255) / 256, 256, 0, stream>>>(x_in, Rmat, xb0, x0);

  // per-layer weight pointers: layers 0,1 = main (d_in 9..18), layer 2 = tail
  const float *We1s[3], *be1s[3], *We2s[3], *be2s[3], *Wcs[3], *bcs[3], *Whs[3], *bhs[3];
  for (int l = 0; l < 2; ++l) {
    We1s[l] = (const float*)d_in[9]  + (size_t)l * 82 * HH;
    be1s[l] = (const float*)d_in[10] + (size_t)l * HH;
    We2s[l] = (const float*)d_in[11] + (size_t)l * HH * HH;
    be2s[l] = (const float*)d_in[12] + (size_t)l * HH;
    Wcs[l]  = (const float*)d_in[13] + (size_t)l * HH;
    bcs[l]  = (const float*)d_in[14] + (size_t)l;
    Whs[l]  = (const float*)d_in[15] + (size_t)l * (FF + HH) * FF;
    bhs[l]  = (const float*)d_in[16] + (size_t)l * FF;
  }
  We1s[2] = (const float*)d_in[19]; be1s[2] = (const float*)d_in[20];
  We2s[2] = (const float*)d_in[21]; be2s[2] = (const float*)d_in[22];
  Wcs[2]  = (const float*)d_in[23]; bcs[2]  = (const float*)d_in[24];
  Whs[2]  = (const float*)d_in[25]; bhs[2]  = (const float*)d_in[26];

  // pack weights to fragment-swizzled f16 + temb
  for (int l = 0; l < 3; ++l) {
    pack_swz_f16<<<(EFP * HH + 255) / 256, 256, 0, stream>>>(
        We1s[l], We1p + (size_t)l * EFP * HH, 82, EFP);
    pack_swz_f16<<<(HH * HH + 255) / 256, 256, 0, stream>>>(
        We2s[l], We2p + (size_t)l * HH * HH, HH, HH);
    const float* wt  = (l < 2) ? (const float*)d_in[17] + (size_t)l * HH : (const float*)d_in[17];
    const float* wbt = (l < 2) ? (const float*)d_in[18] + (size_t)l * HH : (const float*)d_in[18];
    temb_kernel<<<1, 256, 0, stream>>>(t_in, wt, wbt, temb + (size_t)l * BB * HH,
                                       (l < 2) ? 1 : 0);
  }

  // 3 fused layers, ping-pong buffers
  float* xs[4] = {xb0, xb1, xb0, xb1};
  float* hs[4] = {hb0, hb1, hb0, hb1};
  for (int l = 0; l < 3; ++l) {
    egnn_layer<<<BB * SS * NN, 256, 0, stream>>>(
        xs[l], hs[l], xs[l + 1], hs[l + 1], bond, emask,
        We1p + (size_t)l * EFP * HH, We2p + (size_t)l * HH * HH,
        be1s[l], be2s[l], temb + (size_t)l * BB * HH,
        Wcs[l], bcs[l], Whs[l], bhs[l]);
  }

  final_kernel<<<((int)nx + 255) / 256, 256, 0, stream>>>(xs[3], x0, amask,
                                                          (float*)d_out);
}
